// RnnEncoder_87462714016616
// MI455X (gfx1250) — matrix operand order repaired and checked by reading, not compiled, not run on hardware
//
#include <hip/hip_runtime.h>
#include <hip/hip_bf16.h>
#include <stdint.h>

// Problem constants (from the reference)
constexpr int kH  = 512;          // hidden / input dim
constexpr int kB  = 64;           // batch
constexpr int kT  = 512;          // sequence length
constexpr int kC  = 10;           // CTX
constexpr int kOT = kT - 2 * kC;  // output time steps (492)
constexpr int kG  = 3 * kH;       // 1536 gate columns

typedef __attribute__((ext_vector_type(16))) __bf16 v16bf;
typedef __attribute__((ext_vector_type(8)))  float  v8f;

union V16 { uint4 q[2]; v16bf v; };

__device__ __forceinline__ uint16_t f2bf(float f) {
  // round-to-nearest-even fp32 -> bf16
  uint32_t u = __float_as_uint(f);
  uint32_t r = u + 0x7FFFu + ((u >> 16) & 1u);
  return (uint16_t)(r >> 16);
}

// ---- Prep 1: v_in [B][T][H] fp32 -> xbf [T][B][H] bf16 (time-major) ----
__global__ void prep_x(const float* __restrict__ vin, uint16_t* __restrict__ xbf) {
  size_t idx = (size_t)blockIdx.x * blockDim.x + threadIdx.x;   // < T*B*H
  int t   = (int)(idx >> 15);          // B*H = 32768 = 2^15
  int rem = (int)(idx & 32767);
  int b   = rem >> 9;                  // H = 512 = 2^9
  int h   = rem & 511;
  xbf[idx] = f2bf(vin[((size_t)b * kT + t) * kH + h]);
}

// ---- Prep 2: W [3H][H] fp32 -> Wt [3H][H] bf16 (B-matrix friendly) ------
// layout order: 0=Wih_f 1=Whh_f 2=Wih_b 3=Whh_b
__global__ void prep_w(const float* __restrict__ Wih_f, const float* __restrict__ Whh_f,
                       const float* __restrict__ Wih_b, const float* __restrict__ Whh_b,
                       uint16_t* __restrict__ wt) {
  uint32_t idx = blockIdx.x * blockDim.x + threadIdx.x;   // < 4*G*H
  uint32_t per = (uint32_t)kG * kH;                       // 786432
  uint32_t mat = idx / per;
  uint32_t rem = idx % per;
  uint32_t k = rem % kH;
  uint32_t n = rem / kH;
  const float* W = (mat == 0) ? Wih_f : (mat == 1) ? Whh_f : (mat == 2) ? Wih_b : Whh_b;
  wt[idx] = f2bf(W[(size_t)n * kH + k]);
}

// ---- Prep 3: zero h state (fp32, 2 dirs) and bf16 h (2 dirs x 2 parity) --
__global__ void prep_h(float* __restrict__ hstate, uint16_t* __restrict__ hbf) {
  int idx = blockIdx.x * blockDim.x + threadIdx.x;  // < 4*B*H = 131072
  if (idx < 2 * kB * kH) hstate[idx] = 0.0f;
  hbf[idx] = 0;
}

// Load one K-chunk (A 16x32 tile + the 3 gate B 32x16 tiles) = 8 b128 loads.
// arow: per-lane A base (includes mrow*H + kb + kslice offset)
// wlan: per-lane W base (includes (ncolb+lane%16)*kH + kslice + kb)
__device__ __forceinline__ void load_chunk(const uint16_t* __restrict__ arow,
                                           const uint16_t* __restrict__ wlan,
                                           int k0,
                                           V16& a, V16& b0, V16& b1, V16& b2) {
  a.q[0] = *(const uint4*)(arow + k0);
  a.q[1] = *(const uint4*)(arow + k0 + 16);
  const uint16_t* wrow = wlan + k0;
  b0.q[0] = *(const uint4*)(wrow);            b0.q[1] = *(const uint4*)(wrow + 16);
  b1.q[0] = *(const uint4*)(wrow + (size_t)kH * kH);       b1.q[1] = *(const uint4*)(wrow + (size_t)kH * kH + 16);
  b2.q[0] = *(const uint4*)(wrow + (size_t)2 * kH * kH);   b2.q[1] = *(const uint4*)(wrow + (size_t)2 * kH * kH + 16);
}

#define WMMA_BF16(A, Bv, C) \
  __builtin_amdgcn_wmma_f32_16x16x32_bf16(false, (A), false, (Bv), (short)0, (C), false, false)

// ---- One recurrent step, both directions. -------------------------------
// grid = (64, 2), block = 256 (8 waves = 2 tiles x 2 roles x 2 K-slices).
// For each 16x16 h' tile, role 0 computes gi = x_t*Wih^T and role 1
// gh = h*Whh^T; each role is split into two K-slices of 256. Every wave
// runs 8 K-chunks of bf16 WMMA with a 4-deep rotated prefetch whose order
// is pinned by sched_barrier(0). All partial accumulators are merged
// through LDS; one wave per tile applies the GRU nonlinearity + residual.
__global__ __launch_bounds__(256) void gru_step(
    const uint16_t* __restrict__ xbf,   // [T][B][H] bf16
    const float*    __restrict__ vin,   // [B][T][H] fp32 (residual add)
    const uint16_t* __restrict__ wt,    // [4][3H][H] bf16 (W layout, k contiguous)
    const float* __restrict__ bih_f, const float* __restrict__ bhh_f,
    const float* __restrict__ bih_b, const float* __restrict__ bhh_b,
    float*    __restrict__ hstate,      // [2][B][H] fp32 (single-buffered)
    uint16_t* __restrict__ hbf,         // [2][2][B][H] bf16 (parity dbl-buf)
    float*    __restrict__ out,         // [B][OT][2H] fp32
    int t) {
  const int dir  = blockIdx.y;
  const int lane = threadIdx.x & 31;
  const int wv   = threadIdx.x >> 5;        // 0..7
  const int tib  = wv >> 2;                 // tile-in-block 0..1
  const int role = (wv >> 1) & 1;           // 0 = gi (x*Wih), 1 = gh (h*Whh)
  const int ks   = wv & 1;                  // K-slice 0..1 (256 each)
  const int tile = blockIdx.x * 2 + tib;    // 0..127
  const int mb = tile >> 5;                 // batch tile 0..3
  const int nb = tile & 31;                 // hidden-col tile 0..31

  const size_t matsz = (size_t)kH * kG;
  const float* bi = dir ? bih_b : bih_f;
  const float* bh = dir ? bhh_b : bhh_f;
  const int tx = dir ? (kT - 1 - t) : t;    // time index into x for this dir
  const int rp = t & 1;                     // read parity for bf16 h

  float*          hst = hstate + (size_t)dir * (kB * kH);
  const uint16_t* hrd = hbf + ((size_t)(dir * 2) + rp)       * (kB * kH);
  uint16_t*       hwr = hbf + ((size_t)(dir * 2) + (rp ^ 1)) * (kB * kH);

  // Per-role operand selection.
  const uint16_t* W = wt + (size_t)(dir * 2 + role) * matsz;    // Wih or Whh
  const int mrow = mb * 16 + (lane & 15);
  const int kb   = (lane >> 4) * 8;     // A-tile K sub-offset per lane half
  const int ksl  = ks * 256;            // K-slice base
  const uint16_t* arow = ((role == 0)
      ? xbf + ((size_t)tx * kB + mrow) * kH + kb      // A = x_t
      : hrd + (size_t)mrow * kH + kb) + ksl;          // A = h_{t-1}
  const int ncolb = nb * 16;
  const uint16_t* wlan = W + (size_t)(ncolb + (lane & 15)) * kH + ksl + kb; // lane = N col

  v8f c0 = {}, c1 = {}, c2 = {};        // gate accumulators: r, z, n

  // 4-deep rotated software pipeline over this wave's 8 K-chunks of 32,
  // prefetch distance 3, order pinned with hard sched barriers.
  constexpr int NCH = 8;
  V16 a[4], b0[4], b1[4], b2[4];
  load_chunk(arow, wlan, 0,  a[0], b0[0], b1[0], b2[0]);
  load_chunk(arow, wlan, 32, a[1], b0[1], b1[1], b2[1]);
  load_chunk(arow, wlan, 64, a[2], b0[2], b1[2], b2[2]);
#pragma unroll
  for (int i = 0; i < NCH; ++i) {
    const int cur = i & 3;
    const int nxt = (i + 3) & 3;
    if (i + 3 < NCH)
      load_chunk(arow, wlan, (i + 3) * 32, a[nxt], b0[nxt], b1[nxt], b2[nxt]);
    __builtin_amdgcn_sched_barrier(0);   // loads above must stay above
    c0 = WMMA_BF16(a[cur].v, b0[cur].v, c0);
    c1 = WMMA_BF16(a[cur].v, b1[cur].v, c1);
    c2 = WMMA_BF16(a[cur].v, b2[cur].v, c2);
  }

  // Merge all partial accumulators through LDS (identical C/D lane layout).
  __shared__ __align__(32) float red[2][2][2][3][32][8];   // 24 KB
  *(v8f*)&red[tib][role][ks][0][lane][0] = c0;
  *(v8f*)&red[tib][role][ks][1][lane][0] = c1;
  *(v8f*)&red[tib][role][ks][2][lane][0] = c2;
  __syncthreads();

  if ((wv & 3) == 0) {                  // one epilogue wave per tile
    v8f gi0 = *(const v8f*)&red[tib][0][0][0][lane][0];
    v8f gi1 = *(const v8f*)&red[tib][0][0][1][lane][0];
    v8f gi2 = *(const v8f*)&red[tib][0][0][2][lane][0];
    gi0 += *(const v8f*)&red[tib][0][1][0][lane][0];
    gi1 += *(const v8f*)&red[tib][0][1][1][lane][0];
    gi2 += *(const v8f*)&red[tib][0][1][2][lane][0];
    v8f gh0 = *(const v8f*)&red[tib][1][0][0][lane][0];
    v8f gh1 = *(const v8f*)&red[tib][1][0][1][lane][0];
    v8f gh2 = *(const v8f*)&red[tib][1][0][2][lane][0];
    gh0 += *(const v8f*)&red[tib][1][1][0][lane][0];
    gh1 += *(const v8f*)&red[tib][1][1][1][lane][0];
    gh2 += *(const v8f*)&red[tib][1][1][2][lane][0];

    // C/D layout: VGPR e, lane L -> (M = e + (L>=16)*8, N = L&15).
    const int ncol  = ncolb + (lane & 15);
    const int mhalf = (lane >> 4) * 8;
    const float bir = bi[ncol], biz = bi[kH + ncol], bin = bi[2 * kH + ncol];
    const float bhr = bh[ncol], bhz = bh[kH + ncol], bhn = bh[2 * kH + ncol];
    const bool valid = (t >= kC) && (t < kT - kC);

#pragma unroll
    for (int e = 0; e < 8; ++e) {
      const int brow = mb * 16 + mhalf + e;
      const size_t hidx = (size_t)brow * kH + ncol;
      float r = gi0[e] + bir + gh0[e] + bhr;
      r = 1.0f / (1.0f + __expf(-r));
      float z = gi1[e] + biz + gh1[e] + bhz;
      z = 1.0f / (1.0f + __expf(-z));
      float n  = tanhf(gi2[e] + bin + r * (gh2[e] + bhn));
      float hp = hst[hidx];
      float hn = (1.0f - z) * n + z * hp;
      hst[hidx] = hn;
      hwr[hidx] = f2bf(hn);
      if (valid) {
        float xv = vin[((size_t)brow * kT + tx) * kH + ncol];
        out[((size_t)brow * kOT + (t - kC)) * (2 * kH) + dir * kH + ncol] = hn + xv;
      }
    }
  }
}

extern "C" void kernel_launch(void* const* d_in, const int* in_sizes, int n_in,
                              void* d_out, int out_size, void* d_ws, size_t ws_size,
                              hipStream_t stream) {
  const float* vin   = (const float*)d_in[0];
  const float* Wih_f = (const float*)d_in[1];
  const float* Whh_f = (const float*)d_in[2];
  const float* bih_f = (const float*)d_in[3];
  const float* bhh_f = (const float*)d_in[4];
  const float* Wih_b = (const float*)d_in[5];
  const float* Whh_b = (const float*)d_in[6];
  const float* bih_b = (const float*)d_in[7];
  const float* bhh_b = (const float*)d_in[8];
  float* out = (float*)d_out;

  // Workspace layout (~40.4 MB total):
  char* ws = (char*)d_ws;
  uint16_t* xbf = (uint16_t*)ws;                                        // T*B*H bf16     = 32 MB
  uint16_t* wt  = (uint16_t*)(ws + (size_t)33554432);                   // 4*H*3H bf16    = 6 MB
  float* hstate = (float*)(ws + (size_t)33554432 + 6291456);            // 2*B*H fp32     = 256 KB
  uint16_t* hbf = (uint16_t*)(ws + (size_t)33554432 + 6291456 + 262144);// 4*B*H bf16     = 256 KB

  prep_x<<<(kT * kB * kH) / 256, 256, 0, stream>>>(vin, xbf);
  prep_w<<<(4 * kG * kH) / 256, 256, 0, stream>>>(Wih_f, Whh_f, Wih_b, Whh_b, wt);
  prep_h<<<(4 * kB * kH) / 256, 256, 0, stream>>>(hstate, hbf);

  // 512 stream-ordered recurrent steps; each covers fwd+bwd directions.
  for (int t = 0; t < kT; ++t) {
    gru_step<<<dim3(64, 2), 256, 0, stream>>>(xbf, vin, wt,
                                              bih_f, bhh_f, bih_b, bhh_b,
                                              hstate, hbf, out, t);
  }
}
